// MultifractalRecalibration_52853867545145
// MI455X (gfx1250) — matrix operand order repaired and checked by reading, not compiled, not run on hardware
//
#include <hip/hip_runtime.h>
#include <stdint.h>

typedef __attribute__((ext_vector_type(16))) __bf16 v16bf;
typedef __attribute__((ext_vector_type(8)))  float  v8f;

#define IMG 224
#define NCH 64
#define TS  16            // output tile edge
#define IT  32            // input tile edge (TS + 2*8 halo)
#define XS_STRIDE 2064    // bytes per bf16 channel plane (32*32*2 + 16 pad, 16B aligned)
#define XS_OFF    0
#define XS_BYTES  (NCH * XS_STRIDE)          // 132096
#define HS_OFF    XS_BYTES
#define HS_BYTES  (NCH * 1024)               // 65536 : per-channel 16x16 dword pair-staging
#define OUT_OFF   (HS_OFF + HS_BYTES)
#define OUT_STRIDE 65                        // dwords per spatial position (+1 pad: conflict-free)
#define OUT_BYTES (256 * OUT_STRIDE * 4)     // 66560
#define LDS_BYTES (OUT_OFF + OUT_BYTES)      // 264192  (<= 320KB WGP LDS)

// ---------- helpers ----------
__device__ __forceinline__ unsigned fenc(float f) {        // order-preserving uint encode
  unsigned u = __float_as_uint(f);
  return (u & 0x80000000u) ? ~u : (u | 0x80000000u);
}
__device__ __forceinline__ float fdec(unsigned e) {
  return __uint_as_float((e & 0x80000000u) ? (e ^ 0x80000000u) : ~e);
}
__device__ __forceinline__ unsigned bfpack(float a, float b) {
  // native f32->bf16 conversions; backend packs into one dword (v_cvt_pk_bf16_f32 class)
  union { __bf16 h[2]; unsigned u; } p;
  p.h[0] = (__bf16)a;
  p.h[1] = (__bf16)b;
  return p.u;
}

// ---------- kernel A: init min/max slots ----------
__global__ void mm_init_kernel(unsigned* mm) {
  int t = threadIdx.x;
  if (t < 4)      mm[t] = 0xFFFFFFFFu;      // running min (encoded)
  else if (t < 8) mm[t] = 0u;               // running max (encoded)
}

// ---------- kernel B: per-sample min/max ----------
__global__ void mm_reduce_kernel(const float* __restrict__ x, unsigned* mm) {
  __shared__ float smin[256], smax[256];
  const int b   = blockIdx.y;
  const int tid = threadIdx.x;
  long base = (long)b * (IMG * IMG * NCH) + (long)blockIdx.x * 8192 + tid;
  float mn = 3.4e38f, mx = -3.4e38f;
#pragma unroll
  for (int i = 0; i < 32; ++i) {
    float v = x[base + (long)i * 256];
    mn = fminf(mn, v); mx = fmaxf(mx, v);
  }
  smin[tid] = mn; smax[tid] = mx;
  __syncthreads();
  for (int s = 128; s > 0; s >>= 1) {
    if (tid < s) {
      smin[tid] = fminf(smin[tid], smin[tid + s]);
      smax[tid] = fmaxf(smax[tid], smax[tid + s]);
    }
    __syncthreads();
  }
  if (tid == 0) {
    atomicMin(&mm[b],     fenc(smin[0]));
    atomicMax(&mm[4 + b], fenc(smax[0]));
  }
}

// ---------- kernel C: fused multifractal recalibration, one 16x16x64 tile / block ----------
__global__ __launch_bounds__(512)
void mfr_fused_kernel(const float* __restrict__ x,
                      const float* __restrict__ olsw,
                      const float* __restrict__ anchors,
                      const float* __restrict__ widths,
                      const float* __restrict__ bnG,
                      const float* __restrict__ bnB,
                      const float* __restrict__ bnM,
                      const float* __restrict__ bnV,
                      float* __restrict__ out,
                      const unsigned* __restrict__ mm) {
  extern __shared__ char smem[];

  const int tid  = threadIdx.x;
  const int lane = tid & 31;
  const int wave = tid >> 5;       // 16 waves; wave owns channels wave*4 .. wave*4+3
  const int row  = lane & 15;
  const int half = lane >> 4;      // 0: lanes 0-15, 1: lanes 16-31

  const int b   = blockIdx.z;
  const int gh0 = blockIdx.y * TS;
  const int gw0 = blockIdx.x * TS;

  // per-sample affine scale
  const float xmin = fdec(mm[b]);
  const float xmax = fdec(mm[4 + b]);
  const float inv  = __fdividef(1.0f, xmax - xmin + 1e-6f);

  // OLS slope coefficients: alpha = sum_s cc_s * log2(m_s + eps)   (ln2 folded in)
  const float LN2 = 0.6931471805599453f;
  float w0 = olsw[0], w1 = olsw[1], w2 = olsw[2], w3 = olsw[3];
  float l0 = LN2, l1 = 2.f * LN2, l2 = 3.f * LN2, l3 = 4.f * LN2;
  float wsum = w0 + w1 + w2 + w3;
  float lbar = (w0 * l0 + w1 * l1 + w2 * l2 + w3 * l3) / wsum;
  float d0 = l0 - lbar, d1 = l1 - lbar, d2 = l2 - lbar, d3 = l3 - lbar;
  float den = w0 * d0 * d0 + w1 * d1 * d1 + w2 * d2 * d2 + w3 * d3 * d3;
  float cc0 = w0 * d0 / den * LN2, cc1 = w1 * d1 / den * LN2;
  float cc2 = w2 * d2 / den * LN2, cc3 = w3 * d3 / den * LN2;

  // ---- phase 1: fill LDS with bf16 xs, zero halo. float4 over 4 consecutive channels ----
  for (int idx = tid; idx < IT * IT * (NCH / 4); idx += 512) {   // 16384 quads, 32 iters
    int c4 = (idx & 15) * 4;
    int wq = (idx >> 4) & 31;
    int hq = idx >> 9;
    int gh = gh0 - 8 + hq, gw = gw0 - 8 + wq;
    float4 v = make_float4(0.f, 0.f, 0.f, 0.f);
    if (gh >= 0 && gh < IMG && gw >= 0 && gw < IMG)
      v = *(const float4*)(x + (((long)b * IMG + gh) * IMG + gw) * NCH + c4);
    const int sp = (hq * 32 + wq) * 2;
    *(__bf16*)(smem + XS_OFF + (c4 + 0) * XS_STRIDE + sp) = (__bf16)((v.x - xmin) * inv);
    *(__bf16*)(smem + XS_OFF + (c4 + 1) * XS_STRIDE + sp) = (__bf16)((v.y - xmin) * inv);
    *(__bf16*)(smem + XS_OFF + (c4 + 2) * XS_STRIDE + sp) = (__bf16)((v.z - xmin) * inv);
    *(__bf16*)(smem + XS_OFF + (c4 + 3) * XS_STRIDE + sp) = (__bf16)((v.w - xmin) * inv);
  }
  __syncthreads();

  // ---- banded box-filter matrices (constant per lane, kept in VGPRs) ----
  // window for scale r: local cols/rows [P+8-lo, P+8+hi], P = output index
  const int LO[4] = {0, 1, 3, 7};
  const int HI[4] = {1, 2, 4, 8};
  union V16 { v16bf bf; unsigned u[8]; uint4 q[2]; };
  V16 mB[4];   // B-operand: 32x16, M_r[K][N]=1 iff K in window(N)
  V16 mA[4];   // A-operand: 16x32, N_r[M][K]=1 iff K in window(M)
#pragma unroll
  for (int s = 0; s < 4; ++s) {
    const int kmin = row + 8 - LO[s];
    const int kmax = row + 8 + HI[s];
#pragma unroll
    for (int v = 0; v < 8; ++v) {          // B layout: v<4: K=2v(+8 hi-half); v>=4: K=16+2(v-4)(+8)
      int kb = (v < 4 ? 2 * v : 16 + 2 * (v - 4)) + (half ? 8 : 0);
      unsigned lo16 = (kb     >= kmin && kb     <= kmax) ? 0x3F80u : 0u;
      unsigned hi16 = (kb + 1 >= kmin && kb + 1 <= kmax) ? 0x3F80u : 0u;
      mB[s].u[v] = lo16 | (hi16 << 16);
    }
#pragma unroll
    for (int v = 0; v < 8; ++v) {          // A layout: lanes<16: K=2v,2v+1 ; lanes>=16: +16
      int kb = 2 * v + (half ? 16 : 0);
      unsigned lo16 = (kb     >= kmin && kb     <= kmax) ? 0x3F80u : 0u;
      unsigned hi16 = (kb + 1 >= kmin && kb + 1 <= kmax) ? 0x3F80u : 0u;
      mA[s].u[v] = lo16 | (hi16 << 16);
    }
  }

  const v8f zacc = {};

  // ---- phase 2: per-channel WMMA box filters + fused tail ----
#pragma unroll 1
  for (int i = 0; i < 4; ++i) {
    const int c = wave * 4 + i;
    const char* xsb = smem + XS_OFF + c * XS_STRIDE;

    // A operands: X rows (top 0-15, bottom 16-31). Two ds_load_b128 each.
    const uint4* ap = (const uint4*)(xsb + row * 64 + half * 16);
    const uint4* bp = (const uint4*)(xsb + (row + 16) * 64 + half * 16);
    V16 Atop, Abot;
    Atop.q[0] = ap[0]; Atop.q[1] = ap[2];
    Abot.q[0] = bp[0]; Abot.q[1] = bp[2];

    unsigned* hsb = (unsigned*)(smem + HS_OFF + c * 1024);
    v8f ms[4];
#pragma unroll
    for (int s = 0; s < 4; ++s) {
      // H-pass: HS = X * M_r  (32x16 result in two 16x16 WMMAs)
      v8f hsT = __builtin_amdgcn_wmma_f32_16x16x32_bf16(false, Atop.bf, false, mB[s].bf,
                                                        (short)0, zacc, false, false);
      v8f hsB = __builtin_amdgcn_wmma_f32_16x16x32_bf16(false, Abot.bf, false, mB[s].bf,
                                                        (short)0, zacc, false, false);
      // stage D-layout -> B-layout pairs: staging[k2][n], word = (HS[2k2][n], HS[2k2+1][n])
#pragma unroll
      for (int vp = 0; vp < 4; ++vp) {
        hsb[(vp + 4 * half) * 16 + row]     = bfpack(hsT[2 * vp], hsT[2 * vp + 1]);
        hsb[(vp + 4 * half + 8) * 16 + row] = bfpack(hsB[2 * vp], hsB[2 * vp + 1]);
      }
      V16 Bhs;                              // LDS is in-order per wave: safe reload
#pragma unroll
      for (int v = 0; v < 8; ++v) Bhs.u[v] = hsb[(v + 8 * half) * 16 + row];
      // V-pass: MS = N_r * HS
      ms[s] = __builtin_amdgcn_wmma_f32_16x16x32_bf16(false, mA[s].bf, false, Bhs.bf,
                                                      (short)0, zacc, false, false);
    }

    // per-channel scalars (wave-uniform -> scalar loads)
    float bscale = bnG[c] * rsqrtf(bnV[c] + 1e-3f);
    float bbias  = bnB[c] - bnM[c] * bscale;
    float anc[8], wd[8];
#pragma unroll
    for (int k = 0; k < 8; ++k) { anc[k] = anchors[c * 8 + k]; wd[k] = widths[c * 8 + k]; }

    float* ost = (float*)(smem + OUT_OFF);
#pragma unroll
    for (int v = 0; v < 8; ++v) {
      float al = cc0 * __log2f(ms[0][v] + 1e-6f) + cc1 * __log2f(ms[1][v] + 1e-6f) +
                 cc2 * __log2f(ms[2][v] + 1e-6f) + cc3 * __log2f(ms[3][v] + 1e-6f);
      float a = fmaf(al, bscale, bbias);
      float soft = 0.0f;
#pragma unroll
      for (int k = 0; k < 8; ++k)
        soft += fmaxf(0.0f, 1.0f - wd[k] * fabsf(a - anc[k]));
      float sig = __fdividef(1.0f, 1.0f + __expf(-soft));
      int io = v + 8 * half, jo = row;
      ost[(io * 16 + jo) * OUT_STRIDE + c] = sig;   // padded stride: conflict-free
    }
  }
  __syncthreads();

  // ---- phase 3: coalesced b128 epilogue  out = x + sigmoid(soft) ----
  const float* ost = (const float*)(smem + OUT_OFF);
  for (int t = tid; t < 256 * (NCH / 4); t += 512) {   // 4096 quads, 8 iters
    int pos = t >> 4;
    int c4  = (t & 15) * 4;
    int io = pos >> 4, jo = pos & 15;
    long gi = (((long)b * IMG + gh0 + io) * IMG + gw0 + jo) * NCH + c4;
    float4 xv = *(const float4*)(x + gi);
    float4 o;
    o.x = xv.x + ost[pos * OUT_STRIDE + c4 + 0];
    o.y = xv.y + ost[pos * OUT_STRIDE + c4 + 1];
    o.z = xv.z + ost[pos * OUT_STRIDE + c4 + 2];
    o.w = xv.w + ost[pos * OUT_STRIDE + c4 + 3];
    *(float4*)(out + gi) = o;
  }
}

extern "C" void kernel_launch(void* const* d_in, const int* in_sizes, int n_in,
                              void* d_out, int out_size, void* d_ws, size_t ws_size,
                              hipStream_t stream) {
  const float* x       = (const float*)d_in[0];
  const float* olsw    = (const float*)d_in[1];
  const float* anchors = (const float*)d_in[2];
  const float* widths  = (const float*)d_in[3];
  const float* bnG     = (const float*)d_in[4];
  const float* bnB     = (const float*)d_in[5];
  const float* bnM     = (const float*)d_in[6];
  const float* bnV     = (const float*)d_in[7];
  float* out   = (float*)d_out;
  unsigned* mm = (unsigned*)d_ws;

  mm_init_kernel<<<dim3(1), dim3(64), 0, stream>>>(mm);
  mm_reduce_kernel<<<dim3(392, 4), dim3(256), 0, stream>>>(x, mm);
  mfr_fused_kernel<<<dim3(IMG / TS, IMG / TS, 4), dim3(512), LDS_BYTES, stream>>>(
      x, olsw, anchors, widths, bnG, bnB, bnM, bnV, out, mm);
}
